// GNNModel_14353780703343
// MI455X (gfx1250) — matrix operand order repaired and checked by reading, not compiled, not run on hardware
//
#include <hip/hip_runtime.h>
#include <hip/hip_bf16.h>

// ---------------------------------------------------------------------------
// CDNA5 (gfx1250) GCN forward: 6 layers of (GEMM -> self+bias -> edge scatter)
// GEMM: bf16-staged operands + v_wmma_f32_16x16x32_bf16 (fp32 accumulate).
// A/B fragments load as global_load_b128 pairs straight into WMMA layout.
// ---------------------------------------------------------------------------

typedef __attribute__((ext_vector_type(16))) __bf16 v16bf;
typedef __attribute__((ext_vector_type(8)))  float  v8f;

union FragBF {
    v16bf v;
    uint4 q[2];
    unsigned short u[16];
};

__device__ __forceinline__ unsigned short f2bf(float f) {
    unsigned int x = __float_as_uint(f);
    // round-to-nearest-even truncation to bf16
    unsigned int r = x + 0x7FFFu + ((x >> 16) & 1u);
    return (unsigned short)(r >> 16);
}

// ---------------------------------------------------------------------------
// Degree / normalization kernels
// ---------------------------------------------------------------------------
__global__ void zero_f32_kernel(float* __restrict__ p, int n) {
    int i = blockIdx.x * blockDim.x + threadIdx.x;
    if (i < n) p[i] = 0.0f;
}

__global__ void deg_count_kernel(const int* __restrict__ ei, float* __restrict__ deg,
                                 int E) {
    int e = blockIdx.x * blockDim.x + threadIdx.x;
    if (e < E) atomicAdd(&deg[ei[E + e]], 1.0f);   // dst = ei[1][e]
}

__global__ void dinv_ns_kernel(float* __restrict__ deg_to_dinv, float* __restrict__ ns,
                               int n) {
    int i = blockIdx.x * blockDim.x + threadIdx.x;
    if (i < n) {
        float di = rsqrtf(deg_to_dinv[i] + 1.0f);
        deg_to_dinv[i] = di;          // in-place: deg -> dinv
        ns[i] = di * di;
    }
}

__global__ void norm_edge_kernel(const int* __restrict__ ei, const float* __restrict__ dinv,
                                 float* __restrict__ ne, int E) {
    int e = blockIdx.x * blockDim.x + threadIdx.x;
    if (e < E) ne[e] = dinv[ei[e]] * dinv[ei[E + e]];
}

// ---------------------------------------------------------------------------
// Activation + fp32 -> bf16 staging of the GEMM A operand (4 elems / thread)
// ---------------------------------------------------------------------------
__global__ void act_bf16_kernel(const float* __restrict__ h,
                                unsigned short* __restrict__ abf,
                                long long n4, int relu) {
    long long i = ((long long)blockIdx.x * blockDim.x + threadIdx.x);
    if (i >= n4) return;
    float4 v = *(const float4*)(h + i * 4);
    if (relu) {
        v.x = fmaxf(v.x, 0.0f); v.y = fmaxf(v.y, 0.0f);
        v.z = fmaxf(v.z, 0.0f); v.w = fmaxf(v.w, 0.0f);
    }
    uint2 o;
    o.x = (unsigned)f2bf(v.x) | ((unsigned)f2bf(v.y) << 16);
    o.y = (unsigned)f2bf(v.z) | ((unsigned)f2bf(v.w) << 16);
    *(uint2*)(abf + i * 4) = o;
}

// ---------------------------------------------------------------------------
// W [Din, Dout] fp32 row-major -> Wt [Dout, Din] bf16 (column-major for WMMA B)
// ---------------------------------------------------------------------------
__global__ void w_bf16_t_kernel(const float* __restrict__ W,
                                unsigned short* __restrict__ wt,
                                int Din, int Dout) {
    int idx = blockIdx.x * blockDim.x + threadIdx.x;
    if (idx >= Din * Dout) return;
    int col = idx / Din;
    int k   = idx % Din;
    wt[idx] = f2bf(W[(size_t)k * Dout + col]);
}

// ---------------------------------------------------------------------------
// WMMA GEMM: T[N, Dout] = Abf[N, Din](bf16) @ Wt[Dout, Din](bf16)^T
// One 16x16 tile per wave; fully unrolled K loop (DIN compile-time).
// A frag: lane row = lane&15, K pairs {0..7}+8*half and {16..23}+8*half.
// B frag: lane col = lane&15, K pairs {0..7}+8*half and {16..23}+8*half.
// ---------------------------------------------------------------------------
template <int DIN>
__global__ void gemm_wmma_bf16_kernel(const unsigned short* __restrict__ Abf,
                                      const unsigned short* __restrict__ Wt,
                                      float* __restrict__ T,
                                      int n_rows, int Dout) {
    const int lane = threadIdx.x & 31;
    const int wave = blockIdx.x * (blockDim.x >> 5) + (threadIdx.x >> 5);
    const int tiles_n = Dout >> 4;
    const int tiles   = (n_rows >> 4) * tiles_n;
    if (wave >= tiles) return;                 // whole-wave exit: EXEC stays all-1s

    const int tm   = wave / tiles_n;
    const int tn   = wave % tiles_n;
    const int half = lane >> 4;
    const int l    = lane & 15;
    const int row  = tm * 16 + l;              // A row for this lane
    const int col  = tn * 16 + l;              // B column for this lane

    const unsigned short* __restrict__ arow = Abf + (size_t)row * DIN + 8 * half;
    const unsigned short* __restrict__ bcol = Wt  + (size_t)col * DIN + 8 * half;

    v8f acc = {};
#pragma unroll
    for (int kb = 0; kb < DIN; kb += 32) {
        FragBF a, b;
        a.q[0] = *(const uint4*)(arow + kb);        // K = kb+8h   .. +8
        a.q[1] = *(const uint4*)(arow + kb + 16);   // K = kb+16+8h.. +8
        b.q[0] = *(const uint4*)(bcol + kb);        // K = kb+8h   .. +8
        b.q[1] = *(const uint4*)(bcol + kb + 16);   // K = kb+16+8h.. +8
        acc = __builtin_amdgcn_wmma_f32_16x16x32_bf16(
            /*neg_a=*/false, a.v, /*neg_b=*/false, b.v,
            /*c_mod=*/(short)0, acc, /*reuse_a=*/false, /*reuse_b=*/false);
    }

    // C/D layout: VGPR i -> row (i + 8*half), col = lane&15
#pragma unroll
    for (int i = 0; i < 8; ++i)
        T[(size_t)(tm * 16 + i + 8 * half) * Dout + tn * 16 + l] = acc[i];
}

// ---------------------------------------------------------------------------
// out = t * norm_self[row] + bias   (full overwrite; float4 vectorized)
// ---------------------------------------------------------------------------
__global__ void self_bias_kernel(const float* __restrict__ t,
                                 const float* __restrict__ ns,
                                 const float* __restrict__ bias,
                                 float* __restrict__ out,
                                 int n_rows, int Dout) {
    const int g = Dout >> 2;
    long long idx = (long long)blockIdx.x * blockDim.x + threadIdx.x;
    if (idx >= (long long)n_rows * g) return;
    const int row = (int)(idx / g);
    const int c   = (int)(idx % g) * 4;
    const float s = ns[row];
    const float4 v  = *(const float4*)(t + (size_t)row * Dout + c);
    const float4 bb = *(const float4*)(bias + c);
    float4 o;
    o.x = v.x * s + bb.x;
    o.y = v.y * s + bb.y;
    o.z = v.z * s + bb.z;
    o.w = v.w * s + bb.w;
    *(float4*)(out + (size_t)row * Dout + c) = o;
}

// ---------------------------------------------------------------------------
// Edge scatter: out[dst] += t[src] * norm_edge[e]   (float atomics)
// ---------------------------------------------------------------------------
__global__ void scatter_kernel(const float* __restrict__ t,
                               float* __restrict__ out,
                               const int* __restrict__ ei,
                               const float* __restrict__ ne,
                               int E, int Dout) {
    const int g = Dout >> 2;
    long long idx = (long long)blockIdx.x * blockDim.x + threadIdx.x;
    if (idx >= (long long)E * g) return;
    const int e = (int)(idx / g);
    const int c = (int)(idx % g) * 4;
    const int s = ei[e];
    const int d = ei[E + e];
    const float w = ne[e];
    const float4 v = *(const float4*)(t + (size_t)s * Dout + c);
    float* o = out + (size_t)d * Dout + c;
    atomicAdd(o + 0, v.x * w);
    atomicAdd(o + 1, v.y * w);
    atomicAdd(o + 2, v.z * w);
    atomicAdd(o + 3, v.w * w);
}

// ---------------------------------------------------------------------------
// In-place sigmoid on final output
// ---------------------------------------------------------------------------
__global__ void sigmoid_kernel(float* __restrict__ p, int n) {
    int i = blockIdx.x * blockDim.x + threadIdx.x;
    if (i < n) p[i] = 1.0f / (1.0f + __expf(-p[i]));
}

// ---------------------------------------------------------------------------
// Host-side orchestration
// ---------------------------------------------------------------------------
static inline int cdiv_ll(long long a, long long b) { return (int)((a + b - 1) / b); }

extern "C" void kernel_launch(void* const* d_in, const int* in_sizes, int n_in,
                              void* d_out, int out_size, void* d_ws, size_t ws_size,
                              hipStream_t stream) {
    (void)in_sizes; (void)n_in; (void)out_size; (void)ws_size;

    constexpr int N = 50000;
    constexpr int E = 800000;
    constexpr int DMAX = 256;
    static const int DIMS[7] = {128, 64, 128, 256, 128, 64, 64};

    const float* x  = (const float*)d_in[0];
    const int*   ei = (const int*)d_in[1];     // [2, E] flat: src row then dst row

    // workspace layout
    float* ws   = (float*)d_ws;
    float* bufT = ws;                                    // t = h @ W        [N,256] f32
    float* bufH = bufT + (size_t)N * DMAX;               // layer out pre-act[N,256] f32
    float* dinv = bufH + (size_t)N * DMAX;               // deg -> dinv      [N]
    float* ns   = dinv + N;                              // norm_self        [N]
    float* ne   = ns + N;                                // norm_edge        [E]
    unsigned short* abf = (unsigned short*)(ne + E);     // bf16 act(h)      [N,256]
    unsigned short* wbf = abf + (size_t)N * DMAX;        // bf16 W^T         [256,256]

    const int BT = 256;

    // ---- normalization precompute ----
    zero_f32_kernel<<<cdiv_ll(N, BT), BT, 0, stream>>>(dinv, N);
    deg_count_kernel<<<cdiv_ll(E, BT), BT, 0, stream>>>(ei, dinv, E);
    dinv_ns_kernel<<<cdiv_ll(N, BT), BT, 0, stream>>>(dinv, ns, N);
    norm_edge_kernel<<<cdiv_ll(E, BT), BT, 0, stream>>>(ei, dinv, ne, E);

    // ---- layers ----
    const float* hcur = x;
    for (int layer = 0; layer < 6; ++layer) {
        const int Din  = DIMS[layer];
        const int Dout = DIMS[layer + 1];
        const float* Wm   = (const float*)d_in[2 + 2 * layer];
        const float* bias = (const float*)d_in[3 + 2 * layer];
        float* t   = bufT;
        float* out = (layer == 5) ? (float*)d_out : bufH;

        // stage A: activation (ReLU for layers 2..6) + bf16 convert
        const long long n4 = (long long)N * Din / 4;
        act_bf16_kernel<<<cdiv_ll(n4, BT), BT, 0, stream>>>(hcur, abf, n4,
                                                            layer == 0 ? 0 : 1);
        // stage B: W -> bf16 transposed
        w_bf16_t_kernel<<<cdiv_ll((long long)Din * Dout, BT), BT, 0, stream>>>(
            Wm, wbf, Din, Dout);

        // GEMM via WMMA
        const int waves  = (N >> 4) * (Dout >> 4);
        const int blocks = cdiv_ll(waves, BT / 32);
        switch (Din) {
            case 64:
                gemm_wmma_bf16_kernel<64><<<blocks, BT, 0, stream>>>(abf, wbf, t, N, Dout);
                break;
            case 128:
                gemm_wmma_bf16_kernel<128><<<blocks, BT, 0, stream>>>(abf, wbf, t, N, Dout);
                break;
            default:
                gemm_wmma_bf16_kernel<256><<<blocks, BT, 0, stream>>>(abf, wbf, t, N, Dout);
                break;
        }

        // out = t * norm_self + bias
        self_bias_kernel<<<cdiv_ll((long long)N * (Dout >> 2), BT), BT, 0, stream>>>(
            t, ns, bias, out, N, Dout);

        // out[dst] += t[src] * norm_edge
        scatter_kernel<<<cdiv_ll((long long)E * (Dout >> 2), BT), BT, 0, stream>>>(
            t, out, ei, ne, E, Dout);

        hcur = out;
    }

    // final sigmoid in place on d_out (N x 64)
    sigmoid_kernel<<<cdiv_ll((long long)N * 64, BT), BT, 0, stream>>>((float*)d_out, N * 64);
}
